// Upsample_unit_43568148251448
// MI455X (gfx1250) — matrix operand order repaired and checked by reading, not compiled, not run on hardware
//
#include <hip/hip_runtime.h>
#include <math.h>

// ---------------------------------------------------------------------------
// Problem constants (match reference)
// ---------------------------------------------------------------------------
#define BB 8
#define NN 2048
#define MM 2048
#define DD 256
#define DG 512
#define DT 64
#define KT 16
#define UP 4
#define MP 2048           // MERGE_PTS
#define NALL 4096         // M + N
#define SS (MM * KT)      // 32768 neighbor-expanded points per batch
#define UPM (UP * MM)     // 8192

typedef __attribute__((ext_vector_type(8)))  float  v8f;
typedef __attribute__((ext_vector_type(8)))  __bf16 v8bf;
typedef __attribute__((ext_vector_type(16))) __bf16 v16bf;

union Frag { struct { v8bf lo, hi; } p; v16bf v; };
union AccU { v8f v; float f[8]; };

// ---------------------------------------------------------------------------
// WMMA GEMM:  Y[b] = op( W (OC x (IC1+IC2)) * [X1[b] ; X2bcast[b]] + bias + Res )
//   X1: IC1 x NP (row stride NP);  X2: IC2 broadcast scalars per channel.
//   permD != 0 remaps row r -> channel (r%permD), point offset (r/permD)*NP.
// Block: 256 thr = 8 waves (2x4); block tile 32 OC x 256 pts; wave tile 16x64
// (4 accumulators); K-step 32; double-buffered bf16 LDS tiles (X transposed,
// 40-half padded stride -> 16B-aligned ds_load_b128 fragments, bank-clean).
// Pipeline per K-step: barrier -> global loads of tile t+1 (regs) -> all
// fragment ds_loads -> 4 WMMAs -> LDS store of tile t+1. One barrier per step.
// ---------------------------------------------------------------------------
__global__ void __launch_bounds__(256)
gemm_wmma_kernel(const float* __restrict__ W, const float* __restrict__ X1,
                 const float* __restrict__ X2, const float* __restrict__ bias,
                 const float* __restrict__ Res, float* __restrict__ Y,
                 int OC, int IC1, int IC2, int NP,
                 long sX1, long sX2, long sRes, long sY,
                 int ldY, int permD, int relu)
{
  __shared__ __bf16 Wt[2][32 * 40];    // 32 rows x 32 K (padded)
  __shared__ __bf16 Xt[2][256 * 40];   // 256 pts x 32 K, transposed (padded)

  const int tid  = threadIdx.x;
  const int lane = tid & 31, wave = tid >> 5;
  const int wr = wave >> 2, wc = wave & 3;
  const int bz = blockIdx.z;
  const int oc0 = blockIdx.y * 32;
  const int p0  = blockIdx.x * 256;
  const int IC  = IC1 + IC2;

  const float* X1b = X1 + (long)bz * sX1;
  const float* X2b = X2 ? (X2 + (long)bz * sX2) : (const float*)0;

  v8f a0 = {}, a1 = {}, a2 = {}, a3 = {};

  const bool wFastAll  = (oc0 + 32 <= OC) && ((IC & 3) == 0);
  const bool xInBounds = (p0 + 256 <= NP);

  float4 wreg;          // W fast-path staging registers
  float4 xreg[8];       // X fast/bcast staging registers

  // mode: 0 = X interior (fast), 1 = broadcast tile, 2 = generic slow
  auto XMODE = [&](int k0) -> int {
    if (xInBounds && (k0 + 32 <= IC1)) return 0;
    if (xInBounds && k0 >= IC1 && (k0 + 32 <= IC)) return 1;
    return 2;
  };

  // Phase 1: issue global loads for K-tile k0 into registers (fast paths only)
  auto LOAD = [&](int k0) {
    if (wFastAll && (k0 + 32 <= IC)) {
      int r = tid >> 3, kk4 = (tid & 7) << 2;
      wreg = *(const float4*)&W[(long)(oc0 + r) * IC + k0 + kk4];
    }
    int xm = XMODE(k0);
    if (xm == 0) {
      #pragma unroll
      for (int i = 0; i < 8; ++i) {
        int e = tid + i * 256;            // 0..2047
        int kk = e >> 6, p4 = (e & 63) << 2;
        xreg[i] = *(const float4*)&X1b[(long)(k0 + kk) * NP + p0 + p4];
      }
    } else if (xm == 1) {
      #pragma unroll
      for (int i = 0; i < 8; ++i) {
        int e = tid + i * 256;
        int kk = e >> 6;
        xreg[i].x = X2b[k0 + kk - IC1];
      }
    }
  };

  // Phase 2: convert + store K-tile k0 into LDS buffer `buf`
  // (generic slow paths do their guarded loads here too)
  auto STORE = [&](int k0, int buf) {
    __bf16* Wd = Wt[buf];
    __bf16* Xd = Xt[buf];
    if (wFastAll && (k0 + 32 <= IC)) {
      int r = tid >> 3, kk4 = (tid & 7) << 2;
      __bf16* d = &Wd[r * 40 + kk4];
      d[0] = (__bf16)wreg.x; d[1] = (__bf16)wreg.y;
      d[2] = (__bf16)wreg.z; d[3] = (__bf16)wreg.w;
    } else {
      #pragma unroll
      for (int i = 0; i < 4; ++i) {
        int e = tid + i * 256;
        int r = e >> 5, kk = e & 31;
        int gr = oc0 + r, gk = k0 + kk;
        float v = (gr < OC && gk < IC) ? W[(long)gr * IC + gk] : 0.f;
        Wd[r * 40 + kk] = (__bf16)v;
      }
    }
    int xm = XMODE(k0);
    if (xm == 0) {
      #pragma unroll
      for (int i = 0; i < 8; ++i) {
        int e = tid + i * 256;
        int kk = e >> 6, p = (e & 63) << 2;
        Xd[(p + 0) * 40 + kk] = (__bf16)xreg[i].x;
        Xd[(p + 1) * 40 + kk] = (__bf16)xreg[i].y;
        Xd[(p + 2) * 40 + kk] = (__bf16)xreg[i].z;
        Xd[(p + 3) * 40 + kk] = (__bf16)xreg[i].w;
      }
    } else if (xm == 1) {
      #pragma unroll
      for (int i = 0; i < 8; ++i) {
        int e = tid + i * 256;
        int kk = e >> 6, p = (e & 63) << 2;
        __bf16 v = (__bf16)xreg[i].x;
        Xd[(p + 0) * 40 + kk] = v;
        Xd[(p + 1) * 40 + kk] = v;
        Xd[(p + 2) * 40 + kk] = v;
        Xd[(p + 3) * 40 + kk] = v;
      }
    } else {
      int p = tid & 127, kb = tid >> 7;
      #pragma unroll
      for (int half = 0; half < 2; ++half) {
        int pp = p + half * 128;
        #pragma unroll
        for (int i = 0; i < 16; ++i) {
          int kk = kb + i * 2;
          int gk = k0 + kk, gp = p0 + pp;
          float v = 0.f;
          if (gp < NP) {
            if (gk < IC1)     v = X1b[(long)gk * NP + gp];
            else if (gk < IC) v = X2b[gk - IC1];
          }
          Xd[pp * 40 + kk] = (__bf16)v;
        }
      }
    }
  };

  const int m   = lane & 15;
  const int ka  = (lane >> 4) * 16;    // A: 16 contiguous K
  const int kbB = (lane >> 4) * 16;    // B: 16 contiguous K
  const int nk  = (IC + 31) >> 5;

  // ---- software pipeline ----
  LOAD(0);
  STORE(0, 0);
  for (int t = 0; t < nk; ++t) {
    const int cur = t & 1;
    __syncthreads();                       // LDS[cur] ready; prev reads done
    if (t + 1 < nk) LOAD((t + 1) * 32);    // overlap next-tile global loads

    // hoist all fragment loads so ds waits can stagger across the 4 WMMAs
    Frag fa, fb0, fb1, fb2, fb3;
    const __bf16* wrow = &Wt[cur][(wr * 16 + m) * 40];
    fa.p.lo = *(const v8bf*)(wrow + ka);
    fa.p.hi = *(const v8bf*)(wrow + ka + 8);
    const __bf16* xq0 = &Xt[cur][(wc * 64 + 0 * 16 + m) * 40];
    const __bf16* xq1 = &Xt[cur][(wc * 64 + 1 * 16 + m) * 40];
    const __bf16* xq2 = &Xt[cur][(wc * 64 + 2 * 16 + m) * 40];
    const __bf16* xq3 = &Xt[cur][(wc * 64 + 3 * 16 + m) * 40];
    fb0.p.lo = *(const v8bf*)(xq0 + kbB);  fb0.p.hi = *(const v8bf*)(xq0 + kbB + 8);
    fb1.p.lo = *(const v8bf*)(xq1 + kbB);  fb1.p.hi = *(const v8bf*)(xq1 + kbB + 8);
    fb2.p.lo = *(const v8bf*)(xq2 + kbB);  fb2.p.hi = *(const v8bf*)(xq2 + kbB + 8);
    fb3.p.lo = *(const v8bf*)(xq3 + kbB);  fb3.p.hi = *(const v8bf*)(xq3 + kbB + 8);

    a0 = __builtin_amdgcn_wmma_f32_16x16x32_bf16(false, fa.v, false, fb0.v,
                                                 (short)0, a0, false, false);
    a1 = __builtin_amdgcn_wmma_f32_16x16x32_bf16(false, fa.v, false, fb1.v,
                                                 (short)0, a1, false, false);
    a2 = __builtin_amdgcn_wmma_f32_16x16x32_bf16(false, fa.v, false, fb2.v,
                                                 (short)0, a2, false, false);
    a3 = __builtin_amdgcn_wmma_f32_16x16x32_bf16(false, fa.v, false, fb3.v,
                                                 (short)0, a3, false, false);

    if (t + 1 < nk) STORE((t + 1) * 32, cur ^ 1);
  }

  // ---- epilogue ----
  const int n = lane & 15, half = lane >> 4;
#define GT_EPI(ACC, SUBI)                                                      \
  {                                                                            \
    AccU u; u.v = ACC;                                                         \
    int col = p0 + wc * 64 + (SUBI) * 16 + n;                                  \
    if (col < NP) {                                                            \
      _Pragma("unroll")                                                        \
      for (int r8 = 0; r8 < 8; ++r8) {                                         \
        int r = oc0 + wr * 16 + half * 8 + r8;                                 \
        if (r >= OC) continue;                                                 \
        float v = u.f[r8];                                                     \
        if (bias) v += bias[r];                                                \
        if (Res)  v += Res[(long)bz * sRes + (long)r * NP + col];              \
        if (relu) v = fmaxf(v, 0.f);                                           \
        long yi;                                                               \
        if (permD) yi = (long)bz * sY + (long)(r % permD) * ldY                \
                        + (long)(r / permD) * NP + col;                        \
        else       yi = (long)bz * sY + (long)r * ldY + col;                   \
        Y[yi] = v;                                                             \
      }                                                                        \
    }                                                                          \
  }
  GT_EPI(a0, 0) GT_EPI(a1, 1) GT_EPI(a2, 2) GT_EPI(a3, 3)
#undef GT_EPI
}

// ---------------------------------------------------------------------------
// 3-NN of `inputs` against `xyz` + inverse-distance weights (refs in LDS)
// ---------------------------------------------------------------------------
__global__ void __launch_bounds__(256)
knn3_kernel(const float* __restrict__ inputs, const float* __restrict__ xyz,
            int* __restrict__ idx3, float* __restrict__ w3)
{
  __shared__ float sx[MM], sy[MM], sz[MM];
  const int b = blockIdx.x, tid = threadIdx.x;
  for (int j = tid; j < MM; j += 256) {
    const float* p = xyz + ((long)b * MM + j) * 3;
    sx[j] = p[0]; sy[j] = p[1]; sz[j] = p[2];
  }
  __syncthreads();
  for (int q = tid; q < NN; q += 256) {
    const float* pq = inputs + ((long)b * NN + q) * 3;
    float qx = pq[0], qy = pq[1], qz = pq[2];
    float qq = qx * qx + qy * qy + qz * qz;
    float d0 = 3.4e38f, d1 = 3.4e38f, d2 = 3.4e38f;
    int   i0 = 0, i1 = 0, i2 = 0;
    for (int j = 0; j < MM; ++j) {
      float rr = sx[j] * sx[j] + sy[j] * sy[j] + sz[j] * sz[j];
      float d = qq + rr - 2.f * (qx * sx[j] + qy * sy[j] + qz * sz[j]);
      if (d < d0)      { d2 = d1; i2 = i1; d1 = d0; i1 = i0; d0 = d; i0 = j; }
      else if (d < d1) { d2 = d1; i2 = i1; d1 = d; i1 = j; }
      else if (d < d2) { d2 = d; i2 = j; }
    }
    float r0 = 1.f / (fmaxf(d0, 0.f) + 1e-8f);
    float r1 = 1.f / (fmaxf(d1, 0.f) + 1e-8f);
    float r2 = 1.f / (fmaxf(d2, 0.f) + 1e-8f);
    float s = r0 + r1 + r2;
    long o = ((long)b * NN + q) * 3;
    idx3[o] = i0; idx3[o + 1] = i1; idx3[o + 2] = i2;
    w3[o] = r0 / s; w3[o + 1] = r1 / s; w3[o + 2] = r2 / s;
  }
}

// inter[b,c,i] = sum_j w_j * last_fts[b,c,idx_j]
__global__ void interp_kernel(const float* __restrict__ lfts,
                              const int* __restrict__ idx3, const float* __restrict__ w3,
                              float* __restrict__ inter)
{
  int i = blockIdx.x * 256 + threadIdx.x;
  int c = blockIdx.y, b = blockIdx.z;
  long o = ((long)b * NN + i) * 3;
  const float* F = lfts + ((long)b * DD + c) * MM;
  float v = w3[o] * F[idx3[o]] + w3[o + 1] * F[idx3[o + 1]] + w3[o + 2] * F[idx3[o + 2]];
  inter[((long)b * DD + c) * NN + i] = v;
}

__global__ void concat_xyz_kernel(const float* __restrict__ xyz,
                                  const float* __restrict__ inputs,
                                  float* __restrict__ xyz_all)
{
  long i = (long)blockIdx.x * 256 + threadIdx.x;
  if (i >= (long)BB * NALL * 3) return;
  int c = (int)(i % 3);
  long p = (i / 3) % NALL, b = i / (3L * NALL);
  xyz_all[i] = (p < MM) ? xyz[(b * MM + p) * 3 + c]
                        : inputs[(b * NN + (p - MM)) * 3 + c];
}

// ---------------------------------------------------------------------------
// Furthest point sampling: 1 block (1024 thr) per batch; wave32 shfl argmax
// ---------------------------------------------------------------------------
__global__ void __launch_bounds__(1024)
fps_kernel(const float* __restrict__ xyz_all, int* __restrict__ fps_idx)
{
  const int b = blockIdx.x, tid = threadIdx.x;
  const int lane = tid & 31, wave = tid >> 5;
  float px[4], py[4], pz[4], dmin[4];
  #pragma unroll
  for (int r = 0; r < 4; ++r) {
    int p = r * 1024 + tid;
    const float* s = xyz_all + ((long)b * NALL + p) * 3;
    px[r] = s[0]; py[r] = s[1]; pz[r] = s[2]; dmin[r] = 1e10f;
  }
  __shared__ float wval[32];
  __shared__ int   widx[32];
  __shared__ int   sfar;
  int far = 0;
  for (int it = 0; it < MP; ++it) {
    if (tid == 0) fps_idx[(long)b * MP + it] = far;
    const float* c = xyz_all + ((long)b * NALL + far) * 3;
    float fx = c[0], fy = c[1], fz = c[2];
    float bv = -1.f; int bi = 0;
    #pragma unroll
    for (int r = 0; r < 4; ++r) {
      float dx = px[r] - fx, dy = py[r] - fy, dz = pz[r] - fz;
      float d = dx * dx + dy * dy + dz * dz;
      dmin[r] = fminf(dmin[r], d);
      int p = r * 1024 + tid;
      if (dmin[r] > bv || (dmin[r] == bv && p < bi)) { bv = dmin[r]; bi = p; }
    }
    #pragma unroll
    for (int m = 16; m > 0; m >>= 1) {
      float ov = __shfl_xor(bv, m, 32);
      int   oi = __shfl_xor(bi, m, 32);
      if (ov > bv || (ov == bv && oi < bi)) { bv = ov; bi = oi; }
    }
    if (lane == 0) { wval[wave] = bv; widx[wave] = bi; }
    __syncthreads();
    if (wave == 0) {
      bv = wval[lane]; bi = widx[lane];
      #pragma unroll
      for (int m = 16; m > 0; m >>= 1) {
        float ov = __shfl_xor(bv, m, 32);
        int   oi = __shfl_xor(bi, m, 32);
        if (ov > bv || (ov == bv && oi < bi)) { bv = ov; bi = oi; }
      }
      if (lane == 0) sfar = bi;
    }
    __syncthreads();
    far = sfar;
  }
}

__global__ void gather_xyz2_kernel(const float* __restrict__ xyz_all,
                                   const int* __restrict__ fps,
                                   float* __restrict__ xyz2, float* __restrict__ xyz2T)
{
  long i = (long)blockIdx.x * 256 + threadIdx.x;
  if (i >= (long)BB * MP) return;
  long b = i / MP, m = i % MP;
  int s = fps[b * MP + m];
  #pragma unroll
  for (int c = 0; c < 3; ++c) {
    float v = xyz_all[(b * NALL + s) * 3 + c];
    xyz2[(b * MP + m) * 3 + c] = v;
    xyz2T[(b * 3 + c) * MP + m] = v;
  }
}

__global__ void gather_last2_kernel(const float* __restrict__ lfts,
                                    const float* __restrict__ inter,
                                    const int* __restrict__ fps,
                                    float* __restrict__ last2)
{
  long i = (long)blockIdx.x * 256 + threadIdx.x;
  if (i >= (long)BB * DD * MP) return;
  long b = i / ((long)DD * MP);
  long c = (i / MP) % DD, m = i % MP;
  int s = fps[b * MP + m];
  float v = (s < MM) ? lfts[(b * DD + c) * MM + s]
                     : inter[(b * DD + c) * NN + (s - MM)];
  last2[i] = v;
}

// rows 0..127 copy, rows 128..255 broadcast row-max
__global__ void __launch_bounds__(256)
maxcat_kernel(const float* __restrict__ cur1, float* __restrict__ curcat)
{
  __shared__ float sm[256];
  const int c = blockIdx.x, b = blockIdx.y, tid = threadIdx.x;
  const float* src = cur1 + ((long)b * 128 + c) * MP;
  float m = -3.4e38f;
  for (int p = tid; p < MP; p += 256) m = fmaxf(m, src[p]);
  sm[tid] = m; __syncthreads();
  for (int s = 128; s > 0; s >>= 1) {
    if (tid < s) sm[tid] = fmaxf(sm[tid], sm[tid + s]);
    __syncthreads();
  }
  float mx = sm[0];
  float* d0 = curcat + ((long)b * 256 + c) * MP;
  float* d1 = curcat + ((long)b * 256 + 128 + c) * MP;
  for (int p = tid; p < MP; p += 256) { d0[p] = src[p]; d1[p] = mx; }
}

// 16-NN of xyz2 against itself
__global__ void __launch_bounds__(256)
knn16_kernel(const float* __restrict__ xyz2, int* __restrict__ idx16)
{
  __shared__ float sx[MP], sy[MP], sz[MP];
  const int b = blockIdx.x, tid = threadIdx.x;
  for (int j = tid; j < MP; j += 256) {
    const float* p = xyz2 + ((long)b * MP + j) * 3;
    sx[j] = p[0]; sy[j] = p[1]; sz[j] = p[2];
  }
  __syncthreads();
  for (int q = tid; q < MP; q += 256) {
    float qx = sx[q], qy = sy[q], qz = sz[q];
    float qq = qx * qx + qy * qy + qz * qz;
    float bd[KT]; int bi[KT];
    #pragma unroll
    for (int t = 0; t < KT; ++t) { bd[t] = 3.4e38f; bi[t] = 0; }
    for (int j = 0; j < MP; ++j) {
      float rr = sx[j] * sx[j] + sy[j] * sy[j] + sz[j] * sz[j];
      float d = qq + rr - 2.f * (qx * sx[j] + qy * sy[j] + qz * sz[j]);
      if (d < bd[KT - 1]) {
        int pos = KT - 1;
        while (pos > 0 && d < bd[pos - 1]) {
          bd[pos] = bd[pos - 1]; bi[pos] = bi[pos - 1]; --pos;
        }
        bd[pos] = d; bi[pos] = j;
      }
    }
    int* o = idx16 + ((long)b * MP + q) * KT;
    #pragma unroll
    for (int t = 0; t < KT; ++t) o[t] = bi[t];
  }
}

// relative coords (3 x S) for one batch
__global__ void gt_rel_kernel(const float* __restrict__ xyz2,
                              const int* __restrict__ idx16,
                              float* __restrict__ rel, int b)
{
  int s = blockIdx.x * 256 + threadIdx.x;
  if (s >= SS) return;
  int m = s >> 4;
  int j = idx16[(long)b * SS + s];
  const float* pm = xyz2 + ((long)b * MP + m) * 3;
  const float* pj = xyz2 + ((long)b * MP + j) * 3;
  rel[s]          = pm[0] - pj[0];
  rel[SS + s]     = pm[1] - pj[1];
  rel[2 * SS + s] = pm[2] - pj[2];
}

__global__ void gt_gather_kv_kernel(const float* __restrict__ kb,
                                    const float* __restrict__ vb,
                                    const int* __restrict__ idx16,
                                    float* __restrict__ kg, float* __restrict__ vg, int b)
{
  int s = blockIdx.x * 256 + threadIdx.x;
  int t = blockIdx.y;
  if (s >= SS) return;
  int j = idx16[(long)b * SS + s];
  kg[(long)t * SS + s] = kb[((long)b * DT + t) * MP + j];
  vg[(long)t * SS + s] = vb[((long)b * DT + t) * MP + j];
}

// a = q (broadcast over k) - kg + pe
__global__ void gt_a_kernel(const float* __restrict__ qb, const float* __restrict__ kg,
                            const float* __restrict__ pe, float* __restrict__ a, int b)
{
  int s = blockIdx.x * 256 + threadIdx.x;
  int t = blockIdx.y;
  if (s >= SS) return;
  a[(long)t * SS + s] = qb[((long)b * DT + t) * MP + (s >> 4)]
                      - kg[(long)t * SS + s] + pe[(long)t * SS + s];
}

// softmax over 16 neighbors per (t, m), aggregate (vg + pe)
__global__ void gt_softmax_agg_kernel(const float* __restrict__ a2,
                                      const float* __restrict__ vg,
                                      const float* __restrict__ pe,
                                      float* __restrict__ agg, int b)
{
  int g = blockIdx.x * 256 + threadIdx.x;
  if (g >= DT * MP) return;
  int t = g >> 11, m = g & (MP - 1);
  long base = (long)t * SS + (long)m * KT;
  float mx = -3.4e38f;
  #pragma unroll
  for (int k = 0; k < KT; ++k) mx = fmaxf(mx, a2[base + k]);
  float e[KT]; float sum = 0.f;
  #pragma unroll
  for (int k = 0; k < KT; ++k) { e[k] = expf(a2[base + k] - mx); sum += e[k]; }
  float inv = 1.f / sum, acc = 0.f;
  #pragma unroll
  for (int k = 0; k < KT; ++k) acc += (e[k] * inv) * (vg[base + k] + pe[base + k]);
  agg[((long)b * DT + t) * MP + m] = acc;
}

// out_xyz[b,p,c] = delta[b,c,p] + xyz2[b, p mod M, c]
__global__ void final_xyz_kernel(const float* __restrict__ del,
                                 const float* __restrict__ xyz2,
                                 float* __restrict__ out_xyz)
{
  long i = (long)blockIdx.x * 256 + threadIdx.x;
  if (i >= (long)BB * UPM * 3) return;
  int c = (int)(i % 3);
  long p = (i / 3) % UPM, b = i / (3L * UPM);
  out_xyz[i] = del[(b * 3 + c) * UPM + p] + xyz2[(b * MP + (p & (MP - 1))) * 3 + c];
}

// ---------------------------------------------------------------------------
// Host launcher
// ---------------------------------------------------------------------------
static void launch_gemm(hipStream_t st, const float* W, const float* X1, const float* X2,
                        const float* bias, const float* Res, float* Y,
                        int OC, int IC1, int IC2, int NP,
                        long sX1, long sX2, long sRes, long sY,
                        int ldY, int permD, int relu, int nb)
{
  dim3 g((NP + 255) / 256, (OC + 31) / 32, nb);
  gemm_wmma_kernel<<<g, 256, 0, st>>>(W, X1, X2, bias, Res, Y, OC, IC1, IC2, NP,
                                      sX1, sX2, sRes, sY, ldY, permD, relu);
}

extern "C" void kernel_launch(void* const* d_in, const int* in_sizes, int n_in,
                              void* d_out, int out_size, void* d_ws, size_t ws_size,
                              hipStream_t stream)
{
  (void)in_sizes; (void)n_in; (void)out_size; (void)ws_size;
  const float* inputs = (const float*)d_in[0];
  const float* xyz    = (const float*)d_in[1];
  const float* gfts   = (const float*)d_in[2];   // (B, 512, 1)
  const float* lfts   = (const float*)d_in[3];   // (B, 256, 2048)
  // params in setup_inputs() dict insertion order
  const float* e1_W1 = (const float*)d_in[4];  const float* e1_b1 = (const float*)d_in[5];
  const float* e1_W2 = (const float*)d_in[6];  const float* e1_b2 = (const float*)d_in[7];
  const float* e2_W1 = (const float*)d_in[8];  const float* e2_b1 = (const float*)d_in[9];
  const float* e2_W2 = (const float*)d_in[10]; const float* e2_b2 = (const float*)d_in[11];
  const float* o1_W1 = (const float*)d_in[12]; const float* o1_b1 = (const float*)d_in[13];
  const float* o1_W2 = (const float*)d_in[14]; const float* o1_b2 = (const float*)d_in[15];
  const float* o1_Wr = (const float*)d_in[16];
  const float* o2_W1 = (const float*)d_in[17]; const float* o2_b1 = (const float*)d_in[18];
  const float* o2_W2 = (const float*)d_in[19]; const float* o2_b2 = (const float*)d_in[20];
  const float* o3_W1 = (const float*)d_in[21]; const float* o3_b1 = (const float*)d_in[22];
  const float* o3_W2 = (const float*)d_in[23]; const float* o3_b2 = (const float*)d_in[24];
  const float* o3_Wr = (const float*)d_in[25];
  const float* o4_W1 = (const float*)d_in[26]; const float* o4_b1 = (const float*)d_in[27];
  const float* o4_W2 = (const float*)d_in[28]; const float* o4_b2 = (const float*)d_in[29];
  const float* gtW[2][9];   // Wq Wk Wv Wp1 Wp2 Wa1 Wa2 Wo bo  (l3, up)
  for (int g = 0; g < 2; ++g)
    for (int k = 0; k < 9; ++k)
      gtW[g][k] = (const float*)d_in[30 + g * 9 + k];

  // ---- workspace arena ----
  size_t off = 0;
  auto AF = [&](size_t n) { float* p = (float*)((char*)d_ws + off);
                            off += ((n * 4 + 255) & ~(size_t)255); return p; };
  auto AI = [&](size_t n) { int* p = (int*)((char*)d_ws + off);
                            off += ((n * 4 + 255) & ~(size_t)255); return p; };

  float* inter   = AF((size_t)BB * DD * NN);
  float* xyz_all = AF((size_t)BB * NALL * 3);
  float* w3      = AF((size_t)BB * NN * 3);
  int*   idx3    = AI((size_t)BB * NN * 3);
  int*   fps     = AI((size_t)BB * MP);
  float* xyz2    = AF((size_t)BB * MP * 3);
  float* xyz2T   = AF((size_t)BB * 3 * MP);
  float* last2   = AF((size_t)BB * DD * MP);
  float* h64     = AF((size_t)BB * 64 * MP);
  float* cur1    = AF((size_t)BB * 128 * MP);
  float* curcat  = AF((size_t)BB * 256 * MP);
  float* h256    = AF((size_t)BB * 256 * MP);
  float* cur     = AF((size_t)BB * 256 * MP);
  int*   idx16   = AI((size_t)BB * MP * KT);
  float* qb      = AF((size_t)BB * DT * MP);
  float* kb      = AF((size_t)BB * DT * MP);
  float* vb      = AF((size_t)BB * DT * MP);
  float* agg     = AF((size_t)BB * DT * MP);
  float* cur2    = AF((size_t)BB * 256 * MP);
  float* rel     = AF((size_t)3 * SS);         // single batch
  float* kg      = AF((size_t)DT * SS);        // also reused as a2
  float* vg      = AF((size_t)DT * SS);
  float* pe      = AF((size_t)DT * SS);
  float* tmp     = AF((size_t)DT * SS);
  float* ab      = AF((size_t)DT * SS);
  float* YB      = AF((size_t)BB * DD * UPM);  // out_gt / y
  float* HB      = AF((size_t)BB * DD * UPM);
  float* RB      = AF((size_t)BB * DD * UPM);
  float* DEL     = AF((size_t)BB * 3 * UPM);

  float* out_xyz = (float*)d_out;                       // (B, 8192, 3)
  float* out_fts = out_xyz + (size_t)BB * UPM * 3;      // (B, 256, 8192)

  const long sM  = (long)DD * MP;    // 256 x 2048 batch stride
  const long sU  = (long)DD * UPM;   // 256 x 8192 batch stride
  const long sT  = (long)DT * MP;    // 64 x 2048

  // 1) 3-NN interpolation of last_fts onto inputs
  knn3_kernel<<<BB, 256, 0, stream>>>(inputs, xyz, idx3, w3);
  interp_kernel<<<dim3(NN / 256, DD, BB), 256, 0, stream>>>(lfts, idx3, w3, inter);

  // 2) concat coords, FPS, gathers
  {
    long n = (long)BB * NALL * 3;
    concat_xyz_kernel<<<(unsigned)((n + 255) / 256), 256, 0, stream>>>(xyz, inputs, xyz_all);
  }
  fps_kernel<<<BB, 1024, 0, stream>>>(xyz_all, fps);
  {
    long n = (long)BB * MP;
    gather_xyz2_kernel<<<(unsigned)((n + 255) / 256), 256, 0, stream>>>(xyz_all, fps, xyz2, xyz2T);
    n = (long)BB * DD * MP;
    gather_last2_kernel<<<(unsigned)((n + 255) / 256), 256, 0, stream>>>(lfts, inter, fps, last2);
  }

  // 3) displacement feature encoder
  launch_gemm(stream, e1_W1, xyz2T, 0, e1_b1, 0, h64,  64, 3, 0, MP,
              3L * MP, 0, 0, 64L * MP, MP, 0, 1, BB);
  launch_gemm(stream, e1_W2, h64, 0, e1_b2, 0, cur1, 128, 64, 0, MP,
              64L * MP, 0, 0, 128L * MP, MP, 0, 0, BB);
  maxcat_kernel<<<dim3(128, BB), 256, 0, stream>>>(cur1, curcat);
  launch_gemm(stream, e2_W1, curcat, 0, e2_b1, 0, h256, 256, 256, 0, MP,
              sM, 0, 0, sM, MP, 0, 1, BB);
  launch_gemm(stream, e2_W2, h256, 0, e2_b2, 0, cur, 256, 256, 0, MP,
              sM, 0, 0, sM, MP, 0, 0, BB);

  // 4) two group transformers (shared 16-NN graph on xyz2)
  knn16_kernel<<<BB, 256, 0, stream>>>(xyz2, idx16);
  for (int pass = 0; pass < 2; ++pass) {
    const float* Wq = gtW[pass][0]; const float* Wk = gtW[pass][1];
    const float* Wv = gtW[pass][2]; const float* Wp1 = gtW[pass][3];
    const float* Wp2 = gtW[pass][4]; const float* Wa1 = gtW[pass][5];
    const float* Wa2 = gtW[pass][6]; const float* Wo = gtW[pass][7];
    const float* bo = gtW[pass][8];
    const float* curIn = (pass == 0) ? cur : cur2;

    launch_gemm(stream, Wq, curIn, 0, 0, 0, qb, DT, DD, 0, MP, sM, 0, 0, sT, MP, 0, 0, BB);
    launch_gemm(stream, Wk, last2, 0, 0, 0, kb, DT, DD, 0, MP, sM, 0, 0, sT, MP, 0, 0, BB);
    launch_gemm(stream, Wv, last2, 0, 0, 0, vb, DT, DD, 0, MP, sM, 0, 0, sT, MP, 0, 0, BB);

    for (int b = 0; b < BB; ++b) {
      gt_rel_kernel<<<SS / 256, 256, 0, stream>>>(xyz2, idx16, rel, b);
      gt_gather_kv_kernel<<<dim3(SS / 256, DT), 256, 0, stream>>>(kb, vb, idx16, kg, vg, b);
      launch_gemm(stream, Wp1, rel, 0, 0, 0, tmp, DT, 3, 0, SS, 0, 0, 0, 0, SS, 0, 1, 1);
      launch_gemm(stream, Wp2, tmp, 0, 0, 0, pe,  DT, DT, 0, SS, 0, 0, 0, 0, SS, 0, 0, 1);
      gt_a_kernel<<<dim3(SS / 256, DT), 256, 0, stream>>>(qb, kg, pe, ab, b);
      launch_gemm(stream, Wa1, ab,  0, 0, 0, tmp, DT, DT, 0, SS, 0, 0, 0, 0, SS, 0, 1, 1);
      launch_gemm(stream, Wa2, tmp, 0, 0, 0, kg,  DT, DT, 0, SS, 0, 0, 0, 0, SS, 0, 0, 1);
      gt_softmax_agg_kernel<<<(DT * MP) / 256, 256, 0, stream>>>(kg, vg, pe, agg, b);
    }
    if (pass == 0) {
      launch_gemm(stream, Wo, agg, 0, bo, 0, cur2, DD, DT, 0, MP,
                  sT, 0, 0, sM, MP, 0, 0, BB);
    } else {
      // 1024 rows -> (channel, up-block) permuted store into (256 x 8192)
      launch_gemm(stream, Wo, agg, 0, bo, 0, YB, UP * DD, DT, 0, MP,
                  sT, 0, 0, sU, UPM, DD, 0, BB);
    }
  }

  // 5) output head (x = [out | g], g broadcast along points)
  launch_gemm(stream, o1_W1, YB, gfts, o1_b1, 0, HB, DD, DD, DG, UPM,
              sU, DG, 0, sU, UPM, 0, 1, BB);
  launch_gemm(stream, o1_Wr, YB, gfts, 0, 0, RB, DD, DD, DG, UPM,
              sU, DG, 0, sU, UPM, 0, 0, BB);
  launch_gemm(stream, o1_W2, HB, 0, o1_b2, RB, YB, DD, DD, 0, UPM,
              sU, 0, sU, sU, UPM, 0, 0, BB);                 // y -> YB
  launch_gemm(stream, o2_W1, YB, 0, o2_b1, 0, RB, DD, DD, 0, UPM,
              sU, 0, 0, sU, UPM, 0, 1, BB);
  launch_gemm(stream, o2_W2, RB, 0, o2_b2, YB, HB, DD, DD, 0, UPM,
              sU, 0, sU, sU, UPM, 0, 0, BB);                 // y2 -> HB
  launch_gemm(stream, o3_W1, HB, gfts, o3_b1, 0, YB, DD, DD, DG, UPM,
              sU, DG, 0, sU, UPM, 0, 1, BB);
  launch_gemm(stream, o3_Wr, HB, gfts, 0, 0, RB, DD, DD, DG, UPM,
              sU, DG, 0, sU, UPM, 0, 0, BB);
  launch_gemm(stream, o3_W2, YB, 0, o3_b2, RB, out_fts, DD, DD, 0, UPM,
              sU, 0, sU, sU, UPM, 0, 0, BB);                 // out_fts (output #2)
  launch_gemm(stream, o4_W1, out_fts, 0, o4_b1, 0, RB, 64, DD, 0, UPM,
              sU, 0, 0, sU, UPM, 0, 1, BB);
  launch_gemm(stream, o4_W2, RB, 0, o4_b2, 0, DEL, 3, 64, 0, UPM,
              sU, 0, 0, 3L * UPM, UPM, 0, 0, BB);
  {
    long n = (long)BB * UPM * 3;
    final_xyz_kernel<<<(unsigned)((n + 255) / 256), 256, 0, stream>>>(DEL, xyz2, out_xyz);
  }
}
